// attn_head_12902081757780
// MI455X (gfx1250) — matrix operand order repaired and checked by reading, not compiled, not run on hardware
//
#include <hip/hip_runtime.h>
#include <hip/hip_bf16.h>

typedef __attribute__((ext_vector_type(16))) __bf16 v16bf;
typedef __attribute__((ext_vector_type(8)))  float  v8f;

#define NN   8192
#define FIN  512
#define HID  256
#define ALPHA 0.2f

__device__ __forceinline__ float bf2f(unsigned short s) {
    union { unsigned u; float f; } c; c.u = ((unsigned)s) << 16;
    return c.f;
}

// CDNA5 async global->LDS copy (ASYNCcnt-tracked), 16 bytes per lane.
// Generic LDS pointer low 32 bits == LDS byte offset (ISA flat addressing).
__device__ __forceinline__ void async_b128(unsigned lds_off, const void* g) {
    unsigned long long ga = (unsigned long long)(uintptr_t)g;
    asm volatile("global_load_async_to_lds_b128 %0, %1, off"
                 :: "v"(lds_off), "v"(ga) : "memory");
}
__device__ __forceinline__ void wait_async0() {
    asm volatile("s_wait_asynccnt 0x0" ::: "memory");
}

union bfrag { uint4 q[2]; v16bf v; };
union bpack { __bf16 b[8]; uint4 q; };   // 8 f32->bf16 via v_cvt_pk_bf16_f32

// ---------------------------------------------------------------------------
// Kernel 0: Wt[h][k] = bf16(W[k][h])  (pre-transpose+convert, 256 KB)
// ---------------------------------------------------------------------------
__global__ __launch_bounds__(256) void gat_wcvt(const float* __restrict__ W,
                                                unsigned short* __restrict__ Wt)
{
    int idx = blockIdx.x * 256 + threadIdx.x;     // 131072
    int h = idx >> 9, k = idx & 511;
    union { __bf16 b; unsigned short u; } c;
    c.b = (__bf16)W[(size_t)k * HID + h];
    Wt[idx] = c.u;
}

// ---------------------------------------------------------------------------
// Kernel 1: seq_fts = seq @ W  (bf16 WMMA), stored TRANSPOSED bf16 [HID][NN]
// Grid: 512 blocks (16-row i-tiles) x 128 threads (4 waves x 64 h-cols)
// ---------------------------------------------------------------------------
__global__ __launch_bounds__(128) void gat_proj(
    const float* __restrict__ seq, const unsigned short* __restrict__ Wt,
    unsigned short* __restrict__ ftsT)
{
    const int tid = threadIdx.x, wave = tid >> 5, lane = tid & 31;
    const int grp = lane >> 4, lm = lane & 15;
    const int i0 = blockIdx.x * 16;
    const int h0 = wave * 64;

    v8f acc[4] = {};
    for (int k0 = 0; k0 < FIN; k0 += 32) {
        // A fragment: lane's K-elements = 2 contiguous runs of 8 floats
        const float* arow = seq + (size_t)(i0 + lm) * FIN + k0 + grp * 8;
        union { float4 q[4]; float a[16]; } A;
        A.q[0] = *(const float4*)(arow);
        A.q[1] = *(const float4*)(arow + 4);
        A.q[2] = *(const float4*)(arow + 16);
        A.q[3] = *(const float4*)(arow + 20);

        // preload ALL B fragments, then run the WMMA chain densely
        bfrag bu[4];
#pragma unroll
        for (int t = 0; t < 4; ++t) {
            const uint4* wp = (const uint4*)(Wt + (size_t)(h0 + t * 16 + lm) * FIN
                                             + k0 + grp * 8);
            bu[t].q[0] = wp[0];
            bu[t].q[1] = wp[2];
        }

        v16bf a;
#pragma unroll
        for (int e = 0; e < 16; ++e) a[e] = (__bf16)A.a[e];

#pragma unroll
        for (int t = 0; t < 4; ++t)
            acc[t] = __builtin_amdgcn_wmma_f32_16x16x32_bf16(
                false, a, false, bu[t].v, (short)0, acc[t], false, false);
    }
    // C layout: VGPR v -> row M = v + grp*8, col N = lm.
    // Transposed store: row (h0+t*16+lm), cols i0+grp*8..+7 contiguous (one b128).
#pragma unroll
    for (int t = 0; t < 4; ++t) {
        bpack P;
#pragma unroll
        for (int v = 0; v < 8; ++v) P.b[v] = (__bf16)acc[t][v];
        *(uint4*)(ftsT + (size_t)(h0 + t * 16 + lm) * NN + i0 + grp * 8) = P.q;
    }
}

// ---------------------------------------------------------------------------
// Kernel 2: f1[i] = seq_fts[i,:]@a1 + b1 ; f2 likewise
// ---------------------------------------------------------------------------
__global__ __launch_bounds__(256) void gat_f12(
    const unsigned short* __restrict__ ftsT,
    const float* __restrict__ a1, const float* __restrict__ b1,
    const float* __restrict__ a2, const float* __restrict__ b2,
    float* __restrict__ f1, float* __restrict__ f2)
{
    const int i = blockIdx.x * 256 + threadIdx.x;
    float s1 = 0.f, s2 = 0.f;
    for (int h = 0; h < HID; ++h) {
        float v = bf2f(ftsT[(size_t)h * NN + i]);   // coalesced across i
        s1 += v * a1[h];
        s2 += v * a2[h];
    }
    f1[i] = s1 + b1[0];
    f2[i] = s2 + b2[0];
}

// ---------------------------------------------------------------------------
// Kernel 3: maxf2 = max_j f2[j]
// ---------------------------------------------------------------------------
__global__ __launch_bounds__(256) void gat_max(const float* __restrict__ f2,
                                               float* __restrict__ outm)
{
    __shared__ float red[256];
    const int tid = threadIdx.x;
    float m = -3.4e38f;
    for (int j = tid; j < NN; j += 256) m = fmaxf(m, f2[j]);
    red[tid] = m; __syncthreads();
    for (int s = 128; s > 0; s >>= 1) {
        if (tid < s) red[tid] = fmaxf(red[tid], red[tid + s]);
        __syncthreads();
    }
    if (tid == 0) outm[0] = red[0];
}

// ---------------------------------------------------------------------------
// Kernel 4: fused masked-softmax aggregation (flash-style, row max analytic)
// Grid: 128 blocks x 256 threads (8 waves).
// Wave w: i-tile it = w>>1 (16 rows), h-half hh = w&1 (128 cols).
// B tile (32 j x 256 h bf16, 16 KB) double-buffered in LDS via the CDNA5
// async-to-LDS path; next fetch overlaps the current 8-WMMA chain.
// ---------------------------------------------------------------------------
__global__ __launch_bounds__(256) void gat_agg(
    const unsigned short* __restrict__ ftsT,
    const float* __restrict__ f1, const float* __restrict__ f2,
    const float* __restrict__ maxp, const float* __restrict__ bias,
    const float* __restrict__ bz, float* __restrict__ out)
{
    __shared__ unsigned char ldsb[2 * 256 * 64];   // 2 x [h 0..255][32 bf16 j]
    __shared__ float wsum[64];

    const int tid = threadIdx.x, wave = tid >> 5, lane = tid & 31;
    const int grp = lane >> 4, lm = lane & 15;
    const int it = wave >> 1, hh = wave & 1;
    const int i0 = blockIdx.x * 64 + it * 16;
    const int h0 = hh * 128;

    const float mf2 = maxp[0];
    const float f1r = f1[i0 + lm];
    const float tm  = f1r + mf2;
    const float mrow = tm > 0.f ? tm : ALPHA * tm;   // exact row max (bias<=0)
    const float* brow = bias + (size_t)(i0 + lm) * NN;

    const unsigned lbase = (unsigned)(uintptr_t)&ldsb[0];

    // issue tile 0 into buffer 0 (4 x 16B per thread = 16 KB per block)
#pragma unroll
    for (int cc = 0; cc < 4; ++cc) {
        int c = cc * 256 + tid;
        async_b128(lbase + c * 16, ftsT + (size_t)(c >> 2) * NN + (c & 3) * 8);
    }

    v8f acc[8] = {};
    float rs = 0.f;

    for (int t = 0; t < NN / 32; ++t) {
        const int j0 = t * 32;
        const unsigned curoff = (unsigned)(t & 1) * 16384;

        wait_async0();        // own async loads landed
        __syncthreads();      // => all waves' loads landed; prev buffer free

        if (j0 + 32 < NN) {   // prefetch next tile into the other buffer
            const unsigned nxtoff = lbase + (unsigned)((t + 1) & 1) * 16384;
            const unsigned short* src = ftsT + j0 + 32;
#pragma unroll
            for (int cc = 0; cc < 4; ++cc) {
                int c = cc * 256 + tid;
                async_b128(nxtoff + c * 16, src + (size_t)(c >> 2) * NN + (c & 3) * 8);
            }
        }

        // A fragment = softmax weights; K-elements = 2 contiguous runs of 8
        const float* fp = f2 + j0 + grp * 8;
        const float* bp = brow + j0 + grp * 8;
        union { float4 q[4]; float a[16]; } F, Bv;
        F.q[0]  = *(const float4*)(fp);      F.q[1]  = *(const float4*)(fp + 4);
        F.q[2]  = *(const float4*)(fp + 16); F.q[3]  = *(const float4*)(fp + 20);
        Bv.q[0] = *(const float4*)(bp);      Bv.q[1] = *(const float4*)(bp + 4);
        Bv.q[2] = *(const float4*)(bp + 16); Bv.q[3] = *(const float4*)(bp + 20);

        v16bf a;
#pragma unroll
        for (int e = 0; e < 16; ++e) {
            float s  = f1r + F.a[e];
            float lr = s > 0.f ? s : ALPHA * s;
            float w  = __expf(lr - mrow + Bv.a[e]);   // exp(mask bias) folded in
            rs += w;
            a[e] = (__bf16)w;
        }

        // preload all 8 B fragments from LDS, then dense WMMA chain
        bfrag bu[8];
#pragma unroll
        for (int tt = 0; tt < 8; ++tt) {
            const uint4* p = (const uint4*)(ldsb + curoff
                                            + (h0 + tt * 16 + lm) * 64 + grp * 16);
            bu[tt].q[0] = p[0];
            bu[tt].q[1] = p[2];
        }
#pragma unroll
        for (int tt = 0; tt < 8; ++tt)
            acc[tt] = __builtin_amdgcn_wmma_f32_16x16x32_bf16(
                false, a, false, bu[tt].v, (short)0, acc[tt], false, false);
    }

    // lanes m and m+16 hold complementary K-halves of row m -> fold
    float rtot = rs + __shfl_xor(rs, 16, 32);
    __syncthreads();
    if (hh == 0 && lane < 16) wsum[it * 16 + lm] = rtot;
    __syncthreads();

    float inv[8];
#pragma unroll
    for (int v = 0; v < 8; ++v) inv[v] = 1.0f / wsum[it * 16 + grp * 8 + v];

#pragma unroll
    for (int tt = 0; tt < 8; ++tt) {
        const int hc = h0 + tt * 16 + lm;
        const float bzv = bz[hc];
#pragma unroll
        for (int v = 0; v < 8; ++v) {
            float val = acc[tt][v] * inv[v] + bzv;
            float o = val > 0.f ? val : (__expf(val) - 1.f);   // ELU
            out[(size_t)(i0 + grp * 8 + v) * HID + hc] = o;
        }
    }
}

// ---------------------------------------------------------------------------
extern "C" void kernel_launch(void* const* d_in, const int* in_sizes, int n_in,
                              void* d_out, int out_size, void* d_ws, size_t ws_size,
                              hipStream_t stream) {
    const float* seq  = (const float*)d_in[0];
    const float* bias = (const float*)d_in[1];
    const float* W    = (const float*)d_in[2];
    const float* a1   = (const float*)d_in[3];
    const float* b1   = (const float*)d_in[4];
    const float* a2   = (const float*)d_in[5];
    const float* b2   = (const float*)d_in[6];
    const float* bz   = (const float*)d_in[7];
    float* out = (float*)d_out;

    unsigned char* ws = (unsigned char*)d_ws;
    unsigned short* ftsT = (unsigned short*)ws;                 // 4 MB bf16 [HID][NN]
    float* f1v  = (float*)(ws + (size_t)HID * NN * 2);          // 32 KB
    float* f2v  = f1v + NN;                                     // 32 KB
    float* maxm = f2v + NN;                                     // 16 B (padded)
    unsigned short* Wt = (unsigned short*)(maxm + 4);           // 256 KB bf16 [HID][FIN]

    gat_wcvt<<<(FIN * HID) / 256, 256, 0, stream>>>(W, Wt);
    gat_proj<<<NN / 16, 128, 0, stream>>>(seq, Wt, ftsT);
    gat_f12 <<<NN / 256, 256, 0, stream>>>(ftsT, a1, b1, a2, b2, f1v, f2v);
    gat_max <<<1, 256, 0, stream>>>(f2v, maxm);
    gat_agg <<<NN / 64, 256, 0, stream>>>(ftsT, f1v, f2v, maxm, bias, bz, out);
}
